// BitLinear_57629871178271
// MI455X (gfx1250) — matrix operand order repaired and checked by reading, not compiled, not run on hardware
//
#include <hip/hip_runtime.h>

// ---------------------------------------------------------------------------
// BitLinear (BitNet b1.58) forward for MI455X / gfx1250.
//
// out[m,n] = (sum_k bf16(x[m,k]) * ternary(W[n,k])) * scale,  scale = mean|W|
//
// M = N = K = 4096. Compute-bound (AI ~715 FLOP/B) -> WMMA bf16 path.
// Pre-pass converts x -> bf16 and W -> ternary bf16 in workspace; the GEMM
// streams bf16 tiles into double-buffered LDS via async global->LDS loads
// (ASYNCcnt) and overlaps them with v_wmma_f32_16x16x32_bf16 issue.
// ---------------------------------------------------------------------------

typedef __attribute__((ext_vector_type(16))) __bf16 v16bf;
typedef __attribute__((ext_vector_type(8)))  __bf16 v8bf;
typedef __attribute__((ext_vector_type(4)))  __bf16 v4bf;
typedef __attribute__((ext_vector_type(8)))  float  v8f;
typedef __attribute__((ext_vector_type(4)))  float  v4f;
typedef __attribute__((ext_vector_type(4)))  int    v4i;

static constexpr int MDIM = 4096;   // B*S
static constexpr int NDIM = 4096;   // D_OUT
static constexpr int KDIM = 4096;   // D_IN
static constexpr int BM   = 128;
static constexpr int BN   = 128;
static constexpr int BKT  = 64;     // k-elements staged in LDS per stage
static constexpr int LDT  = 72;     // LDS row stride (64 + 8 pad, bank-conflict free)
static constexpr float INV_WCOUNT = 1.0f / 16777216.0f;  // 1/(4096*4096)

#if defined(__AMDGCN__) && __has_builtin(__builtin_amdgcn_global_load_async_to_lds_b128)
#define HAS_ASYNC_LDS 1
#else
#define HAS_ASYNC_LDS 0
#endif

// Async builtin takes typed, address-space-qualified int4 pointers:
//   (v4i addrspace(1)* gaddr, v4i addrspace(3)* ldsaddr, imm offset, imm cpol)
#define AS1V4I(p) ((__attribute__((address_space(1))) v4i*)(uintptr_t)(p))
#define AS3V4I(p) ((__attribute__((address_space(3))) v4i*)(uintptr_t)(p))

template <int N>
__device__ inline void wait_async() {
#if HAS_ASYNC_LDS
#if __has_builtin(__builtin_amdgcn_s_wait_asynccnt)
  __builtin_amdgcn_s_wait_asynccnt(N);
#else
  asm volatile("s_wait_asynccnt %0" ::"n"(N) : "memory");
#endif
#endif
}

// ---------------------------------------------------------------------------
// Pass 1a: per-block partial sums of |W| (4096 blocks -> partials[])
// ---------------------------------------------------------------------------
__global__ __launch_bounds__(256) void absum_partial(const float* __restrict__ w,
                                                     float* __restrict__ partials) {
  __shared__ float red[256];
  const v4f* w4 = (const v4f*)w;
  const int nvec = (NDIM * KDIM) / 4;
  float acc = 0.0f;
  for (int i = blockIdx.x * 256 + threadIdx.x; i < nvec; i += gridDim.x * 256) {
    v4f v = w4[i];
    acc += fabsf(v[0]) + fabsf(v[1]) + fabsf(v[2]) + fabsf(v[3]);
  }
  red[threadIdx.x] = acc;
  __syncthreads();
  for (int s = 128; s > 0; s >>= 1) {
    if (threadIdx.x < s) red[threadIdx.x] += red[threadIdx.x + s];
    __syncthreads();
  }
  if (threadIdx.x == 0) partials[blockIdx.x] = red[0];
}

// Pass 1b: fold 4096 partials -> single sum (deterministic, no atomics)
__global__ __launch_bounds__(256) void absum_final(const float* __restrict__ partials,
                                                   float* __restrict__ sum) {
  __shared__ float red[256];
  float acc = 0.0f;
  for (int i = threadIdx.x; i < 4096; i += 256) acc += partials[i];
  red[threadIdx.x] = acc;
  __syncthreads();
  for (int s = 128; s > 0; s >>= 1) {
    if (threadIdx.x < s) red[threadIdx.x] += red[threadIdx.x + s];
    __syncthreads();
  }
  if (threadIdx.x == 0) sum[0] = red[0];
}

// ---------------------------------------------------------------------------
// Pass 2a: x f32 -> bf16
// ---------------------------------------------------------------------------
__global__ __launch_bounds__(256) void cvt_x_bf16(const float* __restrict__ src,
                                                  void* __restrict__ dstv, int nvec) {
  int i = blockIdx.x * 256 + threadIdx.x;
  if (i >= nvec) return;
  v4f v = ((const v4f*)src)[i];
  v4bf o;
  #pragma unroll
  for (int j = 0; j < 4; ++j) o[j] = (__bf16)v[j];
  ((v4bf*)dstv)[i] = o;
}

// Pass 2b: W f32 -> ternary {-1,0,+1} bf16 (exact in bf16)
__global__ __launch_bounds__(256) void quant_w_bf16(const float* __restrict__ src,
                                                    void* __restrict__ dstv,
                                                    const float* __restrict__ sumPtr,
                                                    int nvec) {
  const float scale = sumPtr[0] * INV_WCOUNT;
  const float qinv  = 1.0f / (scale + 1e-8f);
  int i = blockIdx.x * 256 + threadIdx.x;
  if (i >= nvec) return;
  v4f v = ((const v4f*)src)[i];
  v4bf o;
  #pragma unroll
  for (int j = 0; j < 4; ++j) {
    float q = rintf(fminf(fmaxf(v[j] * qinv, -1.0f), 1.0f));
    o[j] = (__bf16)q;
  }
  ((v4bf*)dstv)[i] = o;
}

// ---------------------------------------------------------------------------
// Global -> LDS tile staging (tile: 128 rows x 64 bf16, LDS row stride LDT).
// PRECONV path: bf16 source, issued as async global->LDS b128 (ASYNCcnt).
// Fallback: f32 source, convert (+quantize) through VGPRs.
// 256 threads, 4 steps, each thread moves 16 B per step.
// ---------------------------------------------------------------------------
template <bool PRECONV, bool QUANT>
__device__ inline void stage_tile(const void* __restrict__ src, int ld,
                                  int rowBase, int kBase,
                                  __bf16* __restrict__ lds, int t, float qinv) {
  #pragma unroll
  for (int step = 0; step < 4; ++step) {
    const int r = step * 32 + (t >> 3);
    const int c = (t & 7) * 8;
    __bf16* dst = lds + r * LDT + c;
    if constexpr (PRECONV) {
      const __bf16* gp = (const __bf16*)src + (size_t)(rowBase + r) * ld + (kBase + c);
#if HAS_ASYNC_LDS
      __builtin_amdgcn_global_load_async_to_lds_b128(AS1V4I(gp), AS3V4I(dst), 0, 0);
#else
      *(v8bf*)dst = *(const v8bf*)gp;
#endif
    } else {
      const float* p = (const float*)src + (size_t)(rowBase + r) * ld + (kBase + c);
      v4f f0 = *(const v4f*)p;
      v4f f1 = *(const v4f*)(p + 4);
      v8bf o;
      #pragma unroll
      for (int j = 0; j < 4; ++j) {
        float a = f0[j], b = f1[j];
        if (QUANT) {
          a = rintf(fminf(fmaxf(a * qinv, -1.0f), 1.0f));
          b = rintf(fminf(fmaxf(b * qinv, -1.0f), 1.0f));
        }
        o[j]     = (__bf16)a;
        o[j + 4] = (__bf16)b;
      }
      *(v8bf*)dst = o;
    }
  }
}

// ---------------------------------------------------------------------------
// WMMA GEMM: C[m,n] = sum_k A[m,k] * B[n,k]   (B = ternary weights, row = n)
// 128x128 block tile, 8 wave32 waves in 2x4 grid -> each wave 64x32.
// Double-buffered LDS; stage i+1 prefetched (async) while stage i computes.
// ---------------------------------------------------------------------------
template <bool PRECONV>
__global__ __launch_bounds__(256) void bitlinear_wmma(const void* __restrict__ Ap,
                                                      const void* __restrict__ Bp,
                                                      float* __restrict__ out,
                                                      const float* __restrict__ sumPtr) {
  __shared__ __align__(16) __bf16 ldsA[2][BM * LDT];
  __shared__ __align__(16) __bf16 ldsB[2][BN * LDT];

  const int t     = threadIdx.x;
  const int lane  = t & 31;
  const int wave  = t >> 5;
  const int waveM = wave & 1;   // 0..1
  const int waveN = wave >> 1;  // 0..3
  const int lr    = lane & 15;
  const int lh    = lane >> 4;

  const float scale = sumPtr[0] * INV_WCOUNT;
  const float qinv  = 1.0f / (scale + 1e-8f);

  const int mBase = blockIdx.x * BM;
  const int nBase = blockIdx.y * BN;

  v8f acc[4][2];
  #pragma unroll
  for (int mi = 0; mi < 4; ++mi)
    #pragma unroll
    for (int ni = 0; ni < 2; ++ni)
      #pragma unroll
      for (int r = 0; r < 8; ++r) acc[mi][ni][r] = 0.0f;

  auto issue_stage = [&](int stageIdx, int buf) {
    const int kt = stageIdx * BKT;
    stage_tile<PRECONV, false>(Ap, KDIM, mBase, kt, ldsA[buf], t, qinv);
    stage_tile<PRECONV, true >(Bp, KDIM, nBase, kt, ldsB[buf], t, qinv);
  };

  constexpr int NSTAGE = KDIM / BKT;  // 64
  issue_stage(0, 0);

  for (int i = 0; i < NSTAGE; ++i) {
    const int cur = i & 1;
    if (i + 1 < NSTAGE) {
      issue_stage(i + 1, cur ^ 1);       // prefetch next stage into other buffer
      wait_async<8>();                   // 8 async ops of stage i+1 may remain in flight
    } else {
      wait_async<0>();
    }
    __syncthreads();                     // stage i visible to all waves

    const __bf16* bufA = ldsA[cur];
    const __bf16* bufB = ldsB[cur];
    #pragma unroll
    for (int k0 = 0; k0 < BKT; k0 += 32) {
      // B fragments: 32x16 bf16; lane half 0 holds K 0..15, half 1 K 16..31
      v16bf bfrag[2];
      #pragma unroll
      for (int ni = 0; ni < 2; ++ni) {
        const __bf16* p = &bufB[(waveN * 32 + ni * 16 + lr) * LDT + k0 + lh * 16];
        v8bf b0 = *(const v8bf*)p;
        v8bf b1 = *(const v8bf*)(p + 8);
        bfrag[ni] = __builtin_shufflevector(b0, b1, 0, 1, 2, 3, 4, 5, 6, 7,
                                            8, 9, 10, 11, 12, 13, 14, 15);
      }
      // A fragments: 16x32 bf16; lane half 0 holds K 0..15, half 1 K 16..31
      #pragma unroll
      for (int mi = 0; mi < 4; ++mi) {
        const __bf16* p = &bufA[(waveM * 64 + mi * 16 + lr) * LDT + k0 + lh * 16];
        v8bf a0 = *(const v8bf*)p;
        v8bf a1 = *(const v8bf*)(p + 8);
        v16bf afrag = __builtin_shufflevector(a0, a1, 0, 1, 2, 3, 4, 5, 6, 7,
                                              8, 9, 10, 11, 12, 13, 14, 15);
        #pragma unroll
        for (int ni = 0; ni < 2; ++ni) {
          acc[mi][ni] = __builtin_amdgcn_wmma_f32_16x16x32_bf16(
              false, afrag, false, bfrag[ni], (short)0, acc[mi][ni], false, false);
        }
      }
    }
    __syncthreads();  // all waves done reading buf `cur` before it is refilled
  }

  // Epilogue: C/D layout: VGPR r, lanes 0-15 -> M=r, N=lane; lanes 16-31 -> M=r+8
  #pragma unroll
  for (int mi = 0; mi < 4; ++mi) {
    #pragma unroll
    for (int ni = 0; ni < 2; ++ni) {
      const int gm0 = mBase + waveM * 64 + mi * 16 + lh * 8;
      const int gn  = nBase + waveN * 32 + ni * 16 + lr;
      float* o = out + (size_t)gm0 * NDIM + gn;
      #pragma unroll
      for (int r = 0; r < 8; ++r) o[(size_t)r * NDIM] = acc[mi][ni][r] * scale;
    }
  }
}

// ---------------------------------------------------------------------------
// Host launcher
// ---------------------------------------------------------------------------
extern "C" void kernel_launch(void* const* d_in, const int* in_sizes, int n_in,
                              void* d_out, int out_size, void* d_ws, size_t ws_size,
                              hipStream_t stream) {
  const float* x = (const float*)d_in[0];  // [2,2048,4096]
  const float* w = (const float*)d_in[1];  // [4096,4096]
  float* out = (float*)d_out;
  char* ws = (char*)d_ws;

  float* sum      = (float*)ws;            // 4 B
  float* partials = (float*)(ws + 64);     // 4096 floats = 16 KB

  constexpr size_t XB     = (size_t)MDIM * KDIM * 2;  // x as bf16: 32 MB
  constexpr size_t WB     = (size_t)NDIM * KDIM * 2;  // W as bf16: 32 MB
  constexpr size_t HDR    = 32768;
  constexpr size_t WSNEED = HDR + XB + WB;

  // Pass 1: scale = mean(|W|)
  absum_partial<<<4096, 256, 0, stream>>>(w, partials);
  absum_final<<<1, 256, 0, stream>>>(partials, sum);

  dim3 grid(MDIM / BM, NDIM / BN);  // 32 x 32 workgroups

  if (ws_size >= WSNEED) {
    void* xb = (void*)(ws + HDR);
    void* wb = (void*)(ws + HDR + XB);
    const int nvec = (MDIM * KDIM) / 4;  // == (NDIM*KDIM)/4
    cvt_x_bf16 <<<nvec / 256, 256, 0, stream>>>(x, xb, nvec);
    quant_w_bf16<<<nvec / 256, 256, 0, stream>>>(w, wb, sum, nvec);
    bitlinear_wmma<true><<<grid, 256, 0, stream>>>(xb, wb, out, sum);
  } else {
    // Fallback: convert + quantize on the fly while staging to LDS
    bitlinear_wmma<false><<<grid, 256, 0, stream>>>((const void*)x, (const void*)w, out, sum);
  }
}
